// Transformer_25185688224076
// MI455X (gfx1250) — matrix operand order repaired and checked by reading, not compiled, not run on hardware
//
#include <hip/hip_runtime.h>

// ---------------------------------------------------------------------------
// CDNA5 (gfx1250) transformer forward.
//  - All matmuls on v_wmma_f32_16x16x32_f16
//  - Double-buffered LDS tiles fed by global_load_async_to_lds_b128 (ASYNCcnt)
// ---------------------------------------------------------------------------

typedef _Float16 h16;
typedef __attribute__((ext_vector_type(16))) _Float16 v16h;
typedef __attribute__((ext_vector_type(8)))  _Float16 v8h;
typedef __attribute__((ext_vector_type(8)))  float    v8f;

static constexpr int Dm    = 512;
static constexpr int Hn    = 8;
static constexpr int DKn   = 64;
static constexpr int DFFn  = 2048;
static constexpr int NRECn = 6;
static constexpr int Vcb   = 32000;
static constexpr int Bn    = 4;
static constexpr int Sn    = 512;
static constexpr int Mrows = Bn * Sn;          // 2048
static constexpr long HS   = (long)Sn * Dm;    // 262144: per-batch stride of [S, H*DK]

// Async global->LDS 16B copy (CDNA5 ISA §15.18 op 98, tracked by ASYNCcnt).
// LDS byte address = low 32 bits of the flat pointer (ISA §10.2 aperture rule).
#define ASYNC_CP16(gsrc, ldst)                                                 \
    asm volatile("global_load_async_to_lds_b128 %0, %1, off"                   \
                 :: "v"((unsigned)(uintptr_t)(ldst)), "v"(gsrc) : "memory")
#define WAIT_ASYNCCNT0() asm volatile("s_wait_asynccnt 0x0" ::: "memory")

// ---------------------------------------------------------------------------
// Generic batched WMMA GEMM:  C = alpha * A(f16) x B(f16) (+bias) (+relu)
//   A[m,k]  at A + m*ldaM + k                (k contiguous)
//   B[k,n]  at B + k*ldbK + n*ldbN
//   C[m,n]  at C + m*ldcM + n                (f32 and/or f16 outputs)
// batch z decomposed as (bz, hz) with z = bz*Hz + hz; per-operand strides.
// Block tile 128x128x32, 256 threads (8 wave32), wave tile 64x32.
// Ping-pong LDS; async DMA staging overlapped with WMMA when tile is full.
// ---------------------------------------------------------------------------
__global__ __launch_bounds__(256)
void k_gemm_f16(const h16* __restrict__ A, const h16* __restrict__ Bm,
                float* __restrict__ Cf, h16* __restrict__ Ch,
                const float* __restrict__ bias,
                int M, int N, int K,
                int ldaM, int ldbK, int ldbN, int ldcM, int Hz,
                long sAb, long sAh, long sBb, long sBh, long sCb, long sCh,
                float alpha, int relu)
{
    __shared__ h16 sA[2][128][48];    // 32 used + pad (16B-aligned rows)
    __shared__ h16 sB[2][32][136];    // 128 used + pad (16B-aligned rows)

    const int tid    = threadIdx.x;
    const int lane   = tid & 31;
    const int wave   = tid >> 5;
    const int m_wave = (wave & 1) * 64;
    const int n_wave = (wave >> 1) * 32;
    const int m0     = blockIdx.y * 128;
    const int n0     = blockIdx.x * 128;

    {   // batch offsets
        const int z  = blockIdx.z;
        const int bz = z / Hz;
        const int hz = z - bz * Hz;
        A  += (long)bz * sAb + (long)hz * sAh;
        Bm += (long)bz * sBb + (long)hz * sBh;
        const long coff = (long)bz * sCb + (long)hz * sCh;
        if (Cf) Cf += coff;
        if (Ch) Ch += coff;
    }

    v8h z8;
#pragma unroll
    for (int q = 0; q < 8; ++q) z8[q] = (h16)0.f;
    const v8f zf = {0.f, 0.f, 0.f, 0.f, 0.f, 0.f, 0.f, 0.f};
    v8f acc[4][2];
#pragma unroll
    for (int i = 0; i < 4; ++i)
#pragma unroll
        for (int j = 0; j < 2; ++j) acc[i][j] = zf;

    // Fully in-bounds, N-contiguous tiles take the async-DMA path.
    const bool fullTile = (m0 + 128 <= M) && (n0 + 128 <= N) && (ldbN == 1);

    auto stage = [&](int k0, int bb) {
        // ---- A tile: 128 x 32 f16 (K always a multiple of 32) ----
#pragma unroll
        for (int e = 0; e < 2; ++e) {
            const int idx = tid + e * 256;       // 0..511
            const int r   = idx >> 2;            // 0..127
            const int c   = (idx & 3) * 8;       // 0,8,16,24
            const int gm  = m0 + r;
            const h16* src = A + (long)gm * ldaM + (k0 + c);
            h16* dst = &sA[bb][r][c];
            if (fullTile) {
                ASYNC_CP16(src, dst);
            } else {
                v8h v = z8;
                if (gm < M) v = *(const v8h*)src;
                *(v8h*)dst = v;
            }
        }
        // ---- B tile: 32 x 128 f16 ----
        if (fullTile) {
#pragma unroll
            for (int e = 0; e < 2; ++e) {
                const int idx = tid + e * 256;   // 0..511
                const int r   = idx >> 4;        // 0..31
                const int c   = (idx & 15) * 8;  // 0..120
                ASYNC_CP16(Bm + (long)(k0 + r) * ldbK + (n0 + c), &sB[bb][r][c]);
            }
        } else if (ldbN == 1) {
#pragma unroll
            for (int e = 0; e < 2; ++e) {
                const int idx = tid + e * 256;
                const int r   = idx >> 4;
                const int c   = (idx & 15) * 8;
                const int gn  = n0 + c;
                v8h v = z8;
                if (gn + 8 <= N) v = *(const v8h*)(Bm + (long)(k0 + r) * ldbK + gn);
                *(v8h*)&sB[bb][r][c] = v;
            }
        } else {
#pragma unroll
            for (int e = 0; e < 16; ++e) {
                const int idx = tid + e * 256;   // 0..4095
                const int r   = idx >> 7;        // 0..31
                const int c   = idx & 127;
                const int gn  = n0 + c;
                h16 v = (h16)0.f;
                if (gn < N) v = Bm[(long)(k0 + r) * ldbK + (long)gn * ldbN];
                sB[bb][r][c] = v;
            }
        }
    };

    stage(0, 0);
    WAIT_ASYNCCNT0();
    __syncthreads();

    const int nk = K >> 5;
    for (int kt = 0; kt < nk; ++kt) {
        const int cur = kt & 1;
        if (kt + 2 < nk) {   // gfx1250 global_prefetch_b8 two tiles ahead
            const int kp = (kt + 2) << 5;
            __builtin_prefetch(A + (long)(m0 + (tid >> 1)) * ldaM + kp, 0, 3);
            __builtin_prefetch(Bm + (long)(kp + (tid >> 3)) * ldbK + (long)n0 * ldbN, 0, 3);
        }
        if (kt + 1 < nk) stage((kt + 1) << 5, cur ^ 1);   // overlap DMA with WMMA

        // ---- fragments per ISA 7.12.2 + WMMA on buffer `cur` ----
        const h16 (*cA)[48]  = sA[cur];
        const h16 (*cB)[136] = sB[cur];
        const int mrow = lane & 15;
        const int kh8  = (lane >> 4) * 8;
        v16h afr[4], bfr[2];
#pragma unroll
        for (int j = 0; j < 2; ++j) {
            const int nn = n_wave + j * 16;
#pragma unroll
            for (int s = 0; s < 8; ++s) {
                bfr[j][s]     = cB[kh8 + s][nn + mrow];
                bfr[j][8 + s] = cB[16 + kh8 + s][nn + mrow];
            }
        }
#pragma unroll
        for (int i = 0; i < 4; ++i) {
            const int m = m_wave + i * 16 + mrow;
            v8h lo = *(const v8h*)&cA[m][kh8];
            v8h hi = *(const v8h*)&cA[m][16 + kh8];
            afr[i] = __builtin_shufflevector(lo, hi, 0,1,2,3,4,5,6,7,8,9,10,11,12,13,14,15);
        }
#pragma unroll
        for (int i = 0; i < 4; ++i)
#pragma unroll
            for (int j = 0; j < 2; ++j)
                acc[i][j] = __builtin_amdgcn_wmma_f32_16x16x32_f16(
                    false, afr[i], false, bfr[j], (short)0, acc[i][j], false, false);

        WAIT_ASYNCCNT0();      // DMA for next tile complete (no-op on sync path)
        __syncthreads();       // all waves done reading `cur` / writing `cur^1`
    }

    // ---- epilogue: C layout VGPR r -> M = r + (lane>>4)*8, N = lane&15 ----
    const int nidx = lane & 15;
    const int mh   = lane >> 4;
#pragma unroll
    for (int i = 0; i < 4; ++i)
#pragma unroll
        for (int j = 0; j < 2; ++j)
#pragma unroll
            for (int r = 0; r < 8; ++r) {
                const int gm = m0 + m_wave + i * 16 + r + mh * 8;
                const int gn = n0 + n_wave + j * 16 + nidx;
                if (gm < M && gn < N) {
                    float v = alpha * acc[i][j][r];
                    if (bias) v += bias[gn];
                    if (relu) v = fmaxf(v, 0.f);
                    if (Cf) Cf[(long)gm * ldcM + gn] = v;
                    if (Ch) Ch[(long)gm * ldcM + gn] = (h16)v;
                }
            }
}

// ---------------------------------------------------------------------------
// Weight packing (f32 -> K-major f16)
// ---------------------------------------------------------------------------
__global__ __launch_bounds__(256)
void k_pack_qkv(const float* __restrict__ w, h16* __restrict__ o)
{   // w: [H=8][D=512][64] -> o: [D=512][H*64=512]
    const int i = blockIdx.x * 256 + threadIdx.x;  // 0..262143
    const int k = i & 63;
    const int d = (i >> 6) & 511;
    const int h = i >> 15;
    o[(long)d * 512 + h * 64 + k] = (h16)w[i];
}

__global__ __launch_bounds__(256)
void k_pack_tr(const float* __restrict__ w, h16* __restrict__ o, int R, int C)
{   // w: [R][C] -> o: [C][R]  (o[c*R + r] = w[r*C + c])
    const long i = (long)blockIdx.x * 256 + threadIdx.x;
    if (i >= (long)R * C) return;
    const int  c = (int)(i % C);
    const long r = i / C;
    o[(long)c * R + r] = (h16)w[i];
}

// ---------------------------------------------------------------------------
// Embedding + positional encoding (faithful bug: PE indexed by BATCH dim)
// ---------------------------------------------------------------------------
__global__ __launch_bounds__(256)
void k_embed_pos(const int* __restrict__ tok, const float* __restrict__ emb,
                 float* __restrict__ x, h16* __restrict__ xh)
{
    const int  bs  = blockIdx.x;   // b*512 + s
    const int  b   = bs >> 9;
    const long t   = (long)tok[bs];
    const float* row = emb + t * Dm;
    for (int d = threadIdx.x; d < Dm; d += 256) {
        const int   i2  = d >> 1;
        const float div = __expf((float)(2 * i2) * (-9.210340371976184f / 512.f));
        const float ang = (float)b * div;
        const float pe  = (d & 1) ? __cosf(ang) : __sinf(ang);
        const float v   = row[d] + pe;
        x [(long)bs * Dm + d] = v;
        xh[(long)bs * Dm + d] = (h16)v;
    }
}

// ---------------------------------------------------------------------------
// Block reduce helper (256 threads, wave32)
// ---------------------------------------------------------------------------
__device__ inline float blk_reduce(float v, float* sh, bool is_max)
{
    for (int o = 16; o > 0; o >>= 1) {
        float t = __shfl_down(v, o, 32);
        v = is_max ? fmaxf(v, t) : v + t;
    }
    const int lane = threadIdx.x & 31, w = threadIdx.x >> 5;
    __syncthreads();
    if (lane == 0) sh[w] = v;
    __syncthreads();
    float r = sh[0];
#pragma unroll
    for (int i = 1; i < 8; ++i) r = is_max ? fmaxf(r, sh[i]) : r + sh[i];
    return r;
}

// ---------------------------------------------------------------------------
// Residual add + LayerNorm (row of 512), dual f32/f16 output
// ---------------------------------------------------------------------------
__global__ __launch_bounds__(256)
void k_ln_add(const float* __restrict__ a, const float* __restrict__ b,
              const float* __restrict__ g, const float* __restrict__ be,
              float* __restrict__ o, h16* __restrict__ oh)
{
    __shared__ float sh[8];
    const long row = blockIdx.x;
    const int  tid = threadIdx.x;
    const float v0 = a[row * Dm + tid]       + b[row * Dm + tid];
    const float v1 = a[row * Dm + tid + 256] + b[row * Dm + tid + 256];
    const float mean = blk_reduce(v0 + v1, sh, false) * (1.f / 512.f);
    const float d0 = v0 - mean, d1 = v1 - mean;
    const float var = blk_reduce(d0 * d0 + d1 * d1, sh, false) * (1.f / 512.f);
    const float inv = rsqrtf(var + 1e-5f);
    const float r0 = d0 * inv * g[tid]       + be[tid];
    const float r1 = d1 * inv * g[tid + 256] + be[tid + 256];
    o [row * Dm + tid]       = r0;
    o [row * Dm + tid + 256] = r1;
    oh[row * Dm + tid]       = (h16)r0;
    oh[row * Dm + tid + 256] = (h16)r1;
}

// ---------------------------------------------------------------------------
// Attention softmax over T=512 (faithful bug: +1.0 added on strict triu)
// grid.x = B*H*S rows; probabilities emitted in f16 for the AV WMMA GEMM
// ---------------------------------------------------------------------------
__global__ __launch_bounds__(256)
void k_softmax_mask(const float* __restrict__ s, h16* __restrict__ p, int causal)
{
    __shared__ float sh[8];
    const long row  = blockIdx.x;
    const int  srow = (int)(row & 511);
    const int  tid  = threadIdx.x;
    float v0 = s[row * 512 + tid];
    float v1 = s[row * 512 + tid + 256];
    if (causal) {
        if (tid > srow)       v0 += 1.f;
        if (tid + 256 > srow) v1 += 1.f;
    }
    const float mx  = blk_reduce(fmaxf(v0, v1), sh, true);
    const float e0  = __expf(v0 - mx), e1 = __expf(v1 - mx);
    const float inv = 1.f / blk_reduce(e0 + e1, sh, false);
    p[row * 512 + tid]       = (h16)(e0 * inv);
    p[row * 512 + tid + 256] = (h16)(e1 * inv);
}

// ---------------------------------------------------------------------------
// Final vocab softmax, in place on d_out rows of 32000
// ---------------------------------------------------------------------------
__global__ __launch_bounds__(256)
void k_softmax_out(float* __restrict__ p)
{
    __shared__ float sh[8];
    float* r = p + (long)blockIdx.x * Vcb;
    const int tid = threadIdx.x;
    float mx = -3.4e38f;
    for (int i = tid; i < Vcb; i += 256) mx = fmaxf(mx, r[i]);
    mx = blk_reduce(mx, sh, true);
    float sm = 0.f;
    for (int i = tid; i < Vcb; i += 256) sm += __expf(r[i] - mx);
    const float inv = 1.f / blk_reduce(sm, sh, false);
    for (int i = tid; i < Vcb; i += 256) r[i] = __expf(r[i] - mx) * inv;
}

// ---------------------------------------------------------------------------
// Host orchestration
// ---------------------------------------------------------------------------
static void launch_gemm(hipStream_t st, const h16* A, const h16* B,
                        float* Cf, h16* Ch, const float* bias,
                        int M, int N, int K,
                        int ldaM, int ldbK, int ldbN, int ldcM,
                        int batches, int Hz,
                        long sAb, long sAh, long sBb, long sBh, long sCb, long sCh,
                        float alpha, int relu)
{
    dim3 grid((N + 127) / 128, (M + 127) / 128, batches);
    k_gemm_f16<<<grid, dim3(256), 0, st>>>(A, B, Cf, Ch, bias, M, N, K,
                                           ldaM, ldbK, ldbN, ldcM, Hz,
                                           sAb, sAh, sBb, sBh, sCb, sCh,
                                           alpha, relu);
}

static inline void* bump(char*& p, size_t bytes)
{
    void* r = p;
    p += (bytes + 255) & ~(size_t)255;
    return r;
}

extern "C" void kernel_launch(void* const* d_in, const int* in_sizes, int n_in,
                              void* d_out, int out_size, void* d_ws, size_t ws_size,
                              hipStream_t stream)
{
    (void)in_sizes; (void)n_in; (void)out_size; (void)ws_size;

    const int*   enc_tok  = (const int*)  d_in[0];
    const int*   dec_tok  = (const int*)  d_in[1];
    const float* emb_W    = (const float*)d_in[2];
    const float* out_b    = (const float*)d_in[3];
    const float* enc_Wq   = (const float*)d_in[4];
    const float* enc_Wk   = (const float*)d_in[5];
    const float* enc_Wv   = (const float*)d_in[6];
    const float* enc_Wo   = (const float*)d_in[7];
    const float* enc_bo   = (const float*)d_in[8];
    const float* enc_W1   = (const float*)d_in[9];
    const float* enc_b1   = (const float*)d_in[10];
    const float* enc_W2   = (const float*)d_in[11];
    const float* enc_b2   = (const float*)d_in[12];
    const float* enc_g    = (const float*)d_in[13];
    const float* enc_be   = (const float*)d_in[14];
    const float* d1_Wq    = (const float*)d_in[15];
    const float* d1_Wk    = (const float*)d_in[16];
    const float* d1_Wv    = (const float*)d_in[17];
    const float* d1_Wo    = (const float*)d_in[18];
    const float* d1_bo    = (const float*)d_in[19];
    const float* d2_Wq    = (const float*)d_in[20];
    const float* d2_Wk    = (const float*)d_in[21];
    const float* d2_Wv    = (const float*)d_in[22];
    const float* d2_Wo    = (const float*)d_in[23];
    const float* d2_bo    = (const float*)d_in[24];
    const float* dd_W1    = (const float*)d_in[25];
    const float* dd_b1    = (const float*)d_in[26];
    const float* dd_W2    = (const float*)d_in[27];
    const float* dd_b2    = (const float*)d_in[28];
    const float* dec_g    = (const float*)d_in[29];
    const float* dec_be   = (const float*)d_in[30];

    // -------- workspace layout (~145 MB) --------
    char* wp = (char*)d_ws;
    const size_t szQKV = (size_t)Dm * Dm * sizeof(h16);         // 512KB
    const size_t szFF  = (size_t)Dm * DFFn * sizeof(h16);       // 2MB
    const size_t szAct = (size_t)Mrows * Dm;                    // elements

    h16* pqe = (h16*)bump(wp, szQKV);  h16* pke = (h16*)bump(wp, szQKV);
    h16* pve = (h16*)bump(wp, szQKV);  h16* poe = (h16*)bump(wp, szQKV);
    h16* pw1e = (h16*)bump(wp, szFF);  h16* pw2e = (h16*)bump(wp, szFF);
    h16* pq1 = (h16*)bump(wp, szQKV);  h16* pk1 = (h16*)bump(wp, szQKV);
    h16* pv1 = (h16*)bump(wp, szQKV);  h16* po1 = (h16*)bump(wp, szQKV);
    h16* pq2 = (h16*)bump(wp, szQKV);  h16* pk2 = (h16*)bump(wp, szQKV);
    h16* pv2 = (h16*)bump(wp, szQKV);  h16* po2 = (h16*)bump(wp, szQKV);
    h16* pw1d = (h16*)bump(wp, szFF);  h16* pw2d = (h16*)bump(wp, szFF);
    h16* pembT = (h16*)bump(wp, (size_t)Dm * Vcb * sizeof(h16));      // [512][32000]

    float* xe   = (float*)bump(wp, szAct * 4);  h16* xe_h = (h16*)bump(wp, szAct * 2);
    float* yd   = (float*)bump(wp, szAct * 4);  h16* yd_h = (h16*)bump(wp, szAct * 2);
    float* x1   = (float*)bump(wp, szAct * 4);  h16* x1_h = (h16*)bump(wp, szAct * 2);
    float* y2   = (float*)bump(wp, szAct * 4);  h16* y2_h = (h16*)bump(wp, szAct * 2);
    float* attn = (float*)bump(wp, szAct * 4);
    h16* Qh = (h16*)bump(wp, szAct * 2);
    h16* Kh = (h16*)bump(wp, szAct * 2);
    h16* Vh = (h16*)bump(wp, szAct * 2);
    h16* Kc = (h16*)bump(wp, szAct * 2);
    h16* Vc = (h16*)bump(wp, szAct * 2);
    h16* oh = (h16*)bump(wp, szAct * 2);
    float* scores = (float*)bump(wp, (size_t)Bn * Hn * Sn * Sn * 4);  // 32MB
    h16*   probs  = (h16*)  bump(wp, (size_t)Bn * Hn * Sn * Sn * 2);  // 16MB
    h16*   ffh    = (h16*)  bump(wp, (size_t)Mrows * DFFn * 2);       // 8MB

    // -------- weight packing --------
    k_pack_qkv<<<1024, 256, 0, stream>>>(enc_Wq, pqe);
    k_pack_qkv<<<1024, 256, 0, stream>>>(enc_Wk, pke);
    k_pack_qkv<<<1024, 256, 0, stream>>>(enc_Wv, pve);
    k_pack_qkv<<<1024, 256, 0, stream>>>(d1_Wq, pq1);
    k_pack_qkv<<<1024, 256, 0, stream>>>(d1_Wk, pk1);
    k_pack_qkv<<<1024, 256, 0, stream>>>(d1_Wv, pv1);
    k_pack_qkv<<<1024, 256, 0, stream>>>(d2_Wq, pq2);
    k_pack_qkv<<<1024, 256, 0, stream>>>(d2_Wk, pk2);
    k_pack_qkv<<<1024, 256, 0, stream>>>(d2_Wv, pv2);
    k_pack_tr<<<1024, 256, 0, stream>>>(enc_Wo, poe, Dm, Dm);
    k_pack_tr<<<1024, 256, 0, stream>>>(d1_Wo, po1, Dm, Dm);
    k_pack_tr<<<1024, 256, 0, stream>>>(d2_Wo, po2, Dm, Dm);
    k_pack_tr<<<4096, 256, 0, stream>>>(enc_W1, pw1e, DFFn, Dm);
    k_pack_tr<<<4096, 256, 0, stream>>>(enc_W2, pw2e, Dm, DFFn);
    k_pack_tr<<<4096, 256, 0, stream>>>(dd_W1, pw1d, DFFn, Dm);
    k_pack_tr<<<4096, 256, 0, stream>>>(dd_W2, pw2d, Dm, DFFn);
    k_pack_tr<<<64000, 256, 0, stream>>>(emb_W, pembT, Vcb, Dm);

    // -------- helpers --------
    auto mha = [&](const h16* qin, const h16* kvin,
                   const h16* PQ, const h16* PK, const h16* PV, const h16* PO,
                   const float* bo, float* outf, int causal,
                   h16* Kbuf, h16* Vbuf, bool computeKV)
    {
        launch_gemm(stream, qin, PQ, nullptr, Qh, nullptr, Mrows, Dm, Dm,
                    Dm, Dm, 1, Dm, 1, 1, 0,0,0,0,0,0, 1.f, 0);
        if (computeKV) {
            launch_gemm(stream, kvin, PK, nullptr, Kbuf, nullptr, Mrows, Dm, Dm,
                        Dm, Dm, 1, Dm, 1, 1, 0,0,0,0,0,0, 1.f, 0);
            launch_gemm(stream, kvin, PV, nullptr, Vbuf, nullptr, Mrows, Dm, Dm,
                        Dm, Dm, 1, Dm, 1, 1, 0,0,0,0,0,0, 1.f, 0);
        }
        // scores[b,h,s,t] = (Q . K) / 8  : per-(b,h) [512x64] x [64x512]
        launch_gemm(stream, Qh, Kbuf, scores, nullptr, nullptr, Sn, Sn, DKn,
                    Dm, 1, Dm, Sn,
                    Bn * Hn, Hn, HS, DKn, HS, DKn, (long)Hn * Sn * Sn, (long)Sn * Sn,
                    0.125f, 0);
        k_softmax_mask<<<Bn * Hn * Sn, 256, 0, stream>>>(scores, probs, causal);
        // o[b,s,h,dv] = probs x V : per-(b,h) [512x512] x [512x64]
        launch_gemm(stream, probs, Vbuf, nullptr, oh, nullptr, Sn, DKn, Sn,
                    Sn, Dm, 1, Dm,
                    Bn * Hn, Hn, (long)Hn * Sn * Sn, (long)Sn * Sn, HS, DKn, HS, DKn,
                    1.f, 0);
        launch_gemm(stream, oh, PO, outf, nullptr, bo, Mrows, Dm, Dm,
                    Dm, Dm, 1, Dm, 1, 1, 0,0,0,0,0,0, 1.f, 0);
    };

    auto ffn = [&](const h16* inh, const h16* W1p, const float* b1,
                   const h16* W2p, const float* b2, float* outf)
    {
        launch_gemm(stream, inh, W1p, nullptr, ffh, b1, Mrows, DFFn, Dm,
                    Dm, DFFn, 1, DFFn, 1, 1, 0,0,0,0,0,0, 1.f, 1);
        launch_gemm(stream, ffh, W2p, outf, nullptr, b2, Mrows, Dm, DFFn,
                    DFFn, Dm, 1, Dm, 1, 1, 0,0,0,0,0,0, 1.f, 0);
    };

    // -------- encoder --------
    k_embed_pos<<<Mrows, 256, 0, stream>>>(enc_tok, emb_W, xe, xe_h);
    for (int l = 0; l < NRECn; ++l) {
        mha(xe_h, xe_h, pqe, pke, pve, poe, enc_bo, attn, 0, Kh, Vh, true);
        k_ln_add<<<Mrows, 256, 0, stream>>>(attn, xe, enc_g, enc_be, x1, x1_h);
        ffn(x1_h, pw1e, enc_b1, pw2e, enc_b2, attn);
        k_ln_add<<<Mrows, 256, 0, stream>>>(x1, attn, enc_g, enc_be, xe, xe_h);
    }
    // xe_h now holds enc_out (f16), untouched by decoder below

    // -------- decoder --------
    k_embed_pos<<<Mrows, 256, 0, stream>>>(dec_tok, emb_W, yd, yd_h);
    // loop-invariant cross-attention K/V from enc_out
    launch_gemm(stream, xe_h, pk2, nullptr, Kc, nullptr, Mrows, Dm, Dm,
                Dm, Dm, 1, Dm, 1, 1, 0,0,0,0,0,0, 1.f, 0);
    launch_gemm(stream, xe_h, pv2, nullptr, Vc, nullptr, Mrows, Dm, Dm,
                Dm, Dm, 1, Dm, 1, 1, 0,0,0,0,0,0, 1.f, 0);
    for (int l = 0; l < NRECn; ++l) {
        mha(yd_h, yd_h, pq1, pk1, pv1, po1, d1_bo, attn, 1, Kh, Vh, true);
        k_ln_add<<<Mrows, 256, 0, stream>>>(yd, attn, dec_g, dec_be, x1, x1_h);
        mha(x1_h, xe_h, pq2, pk2, pv2, po2, d2_bo, attn, 0, Kc, Vc, false);
        k_ln_add<<<Mrows, 256, 0, stream>>>(attn, x1, dec_g, dec_be, y2, y2_h);
        ffn(y2_h, pw1d, dd_b1, pw2d, dd_b2, attn);
        k_ln_add<<<Mrows, 256, 0, stream>>>(attn, y2, dec_g, dec_be, yd, yd_h);
    }

    // -------- logits (tied embedding) + softmax, in place on d_out --------
    launch_gemm(stream, yd_h, pembT, (float*)d_out, nullptr, out_b, Mrows, Vcb, Dm,
                Dm, Vcb, 1, Vcb, 1, 1, 0,0,0,0,0,0, 1.f, 0);
    k_softmax_out<<<Mrows, 256, 0, stream>>>((float*)d_out);
}
